// ViewpointPredictionModel_21354577395800
// MI455X (gfx1250) — matrix operand order repaired and checked by reading, not compiled, not run on hardware
//
#include <hip/hip_runtime.h>
#include <math.h>

// ---------------------------------------------------------------------------
// ViewpointPredictionModel forward for MI455X (gfx1250, wave32, WMMA).
//
// Pipeline (all on `stream`, ordered by stream semantics):
//   prep   : pack Wf/k2/k3 into fp16 K-padded WMMA B-panels
//   enc    : encoder LSTM, block-per-batch-row, recurrence in LDS (VALU)
//   cnn    : per-image fused conv1(VALU)+conv2(WMMA)+conv3(WMMA) in LDS;
//            image staged via GLOBAL_LOAD_ASYNC_TO_LDS_B128 (ASYNCcnt)
//   fc     : 3200x64x8192 GEMM via v_wmma_f32_16x16x32_f16
//   dec    : autoregressive decoder LSTM, block-per-batch-row (VALU)
// ---------------------------------------------------------------------------

typedef __attribute__((ext_vector_type(16))) _Float16 v16h;
typedef __attribute__((ext_vector_type(8)))  _Float16 h8;
typedef __attribute__((ext_vector_type(8)))  float    v8f;

union V16 { v16h v; h8 h[2]; };

#define WMMA_F16(a, b, c) \
  __builtin_amdgcn_wmma_f32_16x16x32_f16(false, (a), false, (b), (short)0, (c), false, false)

__device__ __forceinline__ float sigmoidf_(float x) { return 1.f / (1.f + __expf(-x)); }

// Async copy 16B global -> LDS (no VGPR data path). VDST operand is the raw
// LDS byte offset (flat shared addr[31:0] per CDNA5 aperture mapping).
__device__ __forceinline__ void async_g2l_b128(unsigned lds_off, const void* gptr) {
  asm volatile("global_load_async_to_lds_b128 %0, %1, off"
               :: "v"(lds_off), "v"((unsigned long long)(uintptr_t)gptr)
               : "memory");
}
__device__ __forceinline__ void wait_async0() {
  asm volatile("s_wait_asynccnt 0x0" ::: "memory");
}

// Problem constants
static const int B_   = 128;
static const int M_   = 5;
static const int T_   = 25;
static const int HID_ = 64;
static const int FLAT_ = 8192;

// Workspace layout (bytes, 64B aligned)
static const size_t OFF_WFH  = 0;                               // 64*8192 fp16 = 1 MB
static const size_t OFF_B2   = OFF_WFH + 64 * 8192 * 2;         // 32*192 fp16
static const size_t OFF_B3   = OFF_B2  + 32 * 192 * 2;          // 64*320 fp16
static const size_t OFF_OUT3 = OFF_B3  + 64 * 320 * 2;          // 3200*8192 fp16 = 52.4 MB
static const size_t OFF_SALF = OFF_OUT3 + (size_t)3200 * 8192 * 2; // 3200*64 f32
static const size_t OFF_HENC = OFF_SALF + (size_t)3200 * 64 * 4;   // 128*64 f32
static const size_t OFF_CENC = OFF_HENC + 128 * 64 * 4;            // 128*64 f32

// ---------------------------------------------------------------------------
// prep: fp32 weights -> fp16 WMMA B-panels.
//   WfH : row-major [n=64][k=8192]
//   B2  : row-major [n=32][kpad=192], kk = ci*9+ky*3+kx (im2col order), zero-pad
//   B3  : row-major [n=64][kpad=320]
// ---------------------------------------------------------------------------
__global__ void prep_kernel(const float* __restrict__ Wf,
                            const float* __restrict__ k2,
                            const float* __restrict__ k3,
                            _Float16* __restrict__ WfH,
                            _Float16* __restrict__ B2,
                            _Float16* __restrict__ B3) {
  int stride = gridDim.x * blockDim.x;
  int tid = blockIdx.x * blockDim.x + threadIdx.x;
  for (int i = tid; i < 64 * 8192; i += stride) WfH[i] = (_Float16)Wf[i];
  for (int i = tid; i < 32 * 192; i += stride) {
    int n = i / 192, kk = i % 192;
    B2[i] = (kk < 144) ? (_Float16)k2[n * 144 + kk] : (_Float16)0.f;
  }
  for (int i = tid; i < 64 * 320; i += stride) {
    int n = i / 320, kk = i % 320;
    B3[i] = (kk < 288) ? (_Float16)k3[n * 288 + kk] : (_Float16)0.f;
  }
}

// ---------------------------------------------------------------------------
// Encoder LSTM: one block (64 threads) per batch row, 5 steps looped in LDS.
// ---------------------------------------------------------------------------
__global__ void enc_kernel(const float* __restrict__ pos,
                           const float* __restrict__ Wih,
                           const float* __restrict__ Whh,
                           const float* __restrict__ bih,
                           const float* __restrict__ bhh,
                           float* __restrict__ h_out,
                           float* __restrict__ c_out) {
  int b = blockIdx.x;
  int j = threadIdx.x;          // 0..63 (hidden index)
  __shared__ float h[HID_];
  float c = 0.f;
  h[j] = 0.f;
  __syncthreads();
  for (int t = 0; t < M_; ++t) {
    float x0 = pos[(b * M_ + t) * 3 + 0];
    float x1 = pos[(b * M_ + t) * 3 + 1];
    float x2 = pos[(b * M_ + t) * 3 + 2];
    float g[4];
#pragma unroll
    for (int gi = 0; gi < 4; ++gi) {
      int row = gi * HID_ + j;
      float acc = bih[row] + bhh[row];
      acc += x0 * Wih[row * 3 + 0] + x1 * Wih[row * 3 + 1] + x2 * Wih[row * 3 + 2];
      const float* wr = Whh + row * HID_;
      for (int k = 0; k < HID_; ++k) acc += h[k] * wr[k];
      g[gi] = acc;
    }
    __syncthreads();            // all reads of h done
    c = sigmoidf_(g[1]) * c + sigmoidf_(g[0]) * tanhf(g[2]);
    h[j] = sigmoidf_(g[3]) * tanhf(c);
    __syncthreads();
  }
  h_out[b * HID_ + j] = h[j];
  c_out[b * HID_ + j] = c;
}

// ---------------------------------------------------------------------------
// Implicit-GEMM conv stage via WMMA, operands staged in LDS.
//   inLds  : [Cin][Hin][Win] fp16
//   outLds : [Cout][M] fp16 (M = Hout*Wout), ReLU(bias+acc) applied
//   Bw     : packed fp16 weights [n][Kpad], kk = ci*9+ky*3+kx
// stride-2, pad-1, 3x3. Wout must be a power of two (wShift = log2(Wout)).
// A-fragment per-lane layout: lane half h holds k = 16h + i (contiguous), same order as B.
// ---------------------------------------------------------------------------
__device__ __forceinline__ void conv_wmma(const _Float16* inLds, int Cin, int Hin, int Win,
                                          _Float16* outLds, int Cout, int Hout, int Wout,
                                          int wShift,
                                          const _Float16* __restrict__ Bw,
                                          const float* bias, int Kreal, int Kpad) {
  const int lane  = threadIdx.x & 31;
  const int wave  = threadIdx.x >> 5;
  const int nwave = blockDim.x >> 5;
  const int l15   = lane & 15;
  const int khalf = lane >> 4;
  const int Mdim  = Hout * Wout;
  const int mtiles = Mdim >> 4;
  const int ntiles = Cout >> 4;
  const int ktiles = Kpad >> 5;

  for (int tile = wave; tile < mtiles * ntiles; tile += nwave) {
    const int mt = tile % mtiles;
    const int nt = tile / mtiles;
    const int m  = mt * 16 + l15;
    const int oy = m >> wShift;
    const int ox = m & (Wout - 1);
    v8f acc = {};
    for (int kt = 0; kt < ktiles; ++kt) {
      const int kbase = kt * 32;
      // ---- A fragment: im2col gather from LDS ----
      V16 a;
#pragma unroll
      for (int i = 0; i < 16; ++i) {
        const int kk = kbase + khalf * 16 + i;
        _Float16 val = (_Float16)0.f;
        if (kk < Kreal) {
          const int ci = kk / 9;
          const int r  = kk - ci * 9;
          const int ky = r / 3;
          const int kx = r - ky * 3;
          const int iy = oy * 2 - 1 + ky;
          const int ix = ox * 2 - 1 + kx;
          if ((unsigned)iy < (unsigned)Hin && (unsigned)ix < (unsigned)Win)
            val = inLds[(ci * Hin + iy) * Win + ix];
        }
        a.v[i] = val;
      }
      // ---- B fragment: column n, 16 consecutive K (aligned 32B load) ----
      const _Float16* bp = Bw + (size_t)(nt * 16 + l15) * Kpad + kbase + khalf * 16;
      v16h bfrag = *(const v16h*)bp;
      acc = WMMA_F16(a.v, bfrag, acc);
    }
    // ---- D store: VGPR r -> (M = r + 8*khalf, N = l15) of this tile ----
#pragma unroll
    for (int r = 0; r < 8; ++r) {
      const int mm = mt * 16 + r + 8 * khalf;
      const int nn = nt * 16 + l15;
      float v = acc[r] + bias[nn];
      v = v > 0.f ? v : 0.f;
      outLds[nn * Mdim + mm] = (_Float16)v;
    }
  }
}

// ---------------------------------------------------------------------------
// Fused per-image CNN: conv1 (VALU) -> conv2 (WMMA) -> conv3 (WMMA), all in LDS.
// LDS overlay:
//   img32 (32KB): fp32 image staged by async global->LDS DMA
//   ldsA  (32KB): image fp16 (8192)      -> out2 [32][16][32] (16384)
//   ldsB  (64KB): out1 [16][32][64]      -> out3 [64][8][16]  (8192)
// ---------------------------------------------------------------------------
__global__ __launch_bounds__(256) void cnn_kernel(const float* __restrict__ sal,
                                                  const float* __restrict__ k1,
                                                  const float* __restrict__ bc1,
                                                  const float* __restrict__ bc2,
                                                  const float* __restrict__ bc3,
                                                  const _Float16* __restrict__ B2,
                                                  const _Float16* __restrict__ B3,
                                                  _Float16* __restrict__ out3g) {
  __shared__ float img32[8192];
  __shared__ _Float16 ldsA[16384];
  __shared__ _Float16 ldsB[32768];
  __shared__ float k1w[144];
  __shared__ float b1s[16], b2s[32], b3s[64];

  const int img = blockIdx.x;     // 0..3199 (= b*25 + t)
  const int tid = threadIdx.x;    // 256 threads = 8 waves

  // ---- async-DMA the fp32 image straight into LDS (no VGPR data path) ----
  const float* ip = sal + (size_t)img * 8192;
#pragma unroll
  for (int i = tid; i < 2048; i += 256)   // 2048 x 16B = 32KB
    async_g2l_b128((unsigned)(uintptr_t)&img32[i * 4], ip + i * 4);

  if (tid < 144) k1w[tid] = k1[tid];
  if (tid < 16)  b1s[tid] = bc1[tid];
  if (tid < 32)  b2s[tid] = bc2[tid];
  if (tid < 64)  b3s[tid] = bc3[tid];

  wait_async0();                 // drain this wave's ASYNCcnt
  __syncthreads();               // all waves' DMA visible

  // fp32 -> fp16 conversion in LDS
  for (int i = tid; i < 8192; i += 256) ldsA[i] = (_Float16)img32[i];
  __syncthreads();

  // conv1: 1->16, (64,128)->(32,64), K=9 (VALU)
  for (int idx = tid; idx < 32768; idx += 256) {
    const int cch = idx >> 11;
    const int rem = idx & 2047;
    const int y = rem >> 6;
    const int x = rem & 63;
    float acc = b1s[cch];
    const float* w = k1w + cch * 9;
#pragma unroll
    for (int ky = 0; ky < 3; ++ky) {
      const int iy = 2 * y - 1 + ky;
      if ((unsigned)iy >= 64u) continue;
#pragma unroll
      for (int kx = 0; kx < 3; ++kx) {
        const int ix = 2 * x - 1 + kx;
        if ((unsigned)ix >= 128u) continue;
        acc += (float)ldsA[iy * 128 + ix] * w[ky * 3 + kx];
      }
    }
    ldsB[idx] = (_Float16)(acc > 0.f ? acc : 0.f);
  }
  __syncthreads();

  // conv2: 16->32, (32,64)->(16,32), K=144 pad 192 (WMMA)
  conv_wmma(ldsB, 16, 32, 64, ldsA, 32, 16, 32, 5, B2, b2s, 144, 192);
  __syncthreads();

  // conv3: 32->64, (16,32)->(8,16), K=288 pad 320 (WMMA)
  conv_wmma(ldsA, 32, 16, 32, ldsB, 64, 8, 16, 4, B3, b3s, 288, 320);
  __syncthreads();

  // out3 layout c*128 + y*16 + x matches reference flatten order
  _Float16* op = out3g + (size_t)img * 8192;
  for (int i = tid; i < 8192; i += 256) op[i] = ldsB[i];
}

// ---------------------------------------------------------------------------
// FC: salf[img][n] = out3[img][:] . Wf[n][:] + bf[n]
// GEMM 3200 x 64 x 8192. Block = 128 threads (4 waves); wave w owns the
// 16x16 tile at N = 16*w; 256 k-steps of v_wmma_f32_16x16x32_f16.
// ---------------------------------------------------------------------------
__global__ __launch_bounds__(128) void fc_kernel(const _Float16* __restrict__ out3,
                                                 const _Float16* __restrict__ WfH,
                                                 const float* __restrict__ bf,
                                                 float* __restrict__ salf) {
  const int imgt  = blockIdx.x;          // 0..199 (16 images each)
  const int wave  = threadIdx.x >> 5;    // 0..3 -> N tile
  const int lane  = threadIdx.x & 31;
  const int l15   = lane & 15;
  const int khalf = lane >> 4;

  const _Float16* Abase = out3 + (size_t)(imgt * 16 + l15) * FLAT_; // row = image
  const _Float16* Bbase = WfH + (size_t)(wave * 16 + l15) * FLAT_;  // row = out ch

  v8f acc = {};
  for (int k = 0; k < FLAT_; k += 32) {
    __builtin_prefetch(Abase + k + 512, 0, 0);   // global_prefetch_b8
    V16 a;
    a.v = *(const v16h*)(Abase + k + khalf * 16);        // K 0-15 / 16-31
    v16h bfrag = *(const v16h*)(Bbase + k + khalf * 16); // K 0-15 / 16-31
    acc = WMMA_F16(a.v, bfrag, acc);
  }
  const int nn = wave * 16 + l15;
  const float bias = bf[nn];
#pragma unroll
  for (int r = 0; r < 8; ++r) {
    const int im = imgt * 16 + r + 8 * khalf;
    salf[(size_t)im * HID_ + nn] = acc[r] + bias;
  }
}

// ---------------------------------------------------------------------------
// Decoder: one block (64 threads) per batch row; 25 autoregressive steps.
// ---------------------------------------------------------------------------
__global__ void dec_kernel(const float* __restrict__ pos,
                           const float* __restrict__ salf,
                           const float* __restrict__ W1, const float* __restrict__ b1,
                           const float* __restrict__ W2, const float* __restrict__ b2,
                           const float* __restrict__ Wih, const float* __restrict__ Whh,
                           const float* __restrict__ bih, const float* __restrict__ bhh,
                           const float* __restrict__ Wo, const float* __restrict__ bo,
                           const float* __restrict__ h_in, const float* __restrict__ c_in,
                           float* __restrict__ out) {
  const int b = blockIdx.x;
  const int j = threadIdx.x;    // 0..63
  __shared__ float h[HID_], inp[2 * HID_], hid1[2 * HID_], prev[3], newp[3];
  float c = c_in[b * HID_ + j];
  h[j] = h_in[b * HID_ + j];
  if (j < 3) prev[j] = pos[(b * M_ + (M_ - 1)) * 3 + j];
  __syncthreads();

  for (int t = 0; t < T_; ++t) {
    // sense_pos MLP: hid1 = relu(W1 @ prev + b1)  (128)
    for (int e = j; e < 2 * HID_; e += HID_) {
      float a = b1[e] + W1[e * 3 + 0] * prev[0] + W1[e * 3 + 1] * prev[1] +
                W1[e * 3 + 2] * prev[2];
      hid1[e] = a > 0.f ? a : 0.f;
    }
    __syncthreads();
    // sp = W2 @ hid1 + b2 (64); inp = [sp, salf_t]
    {
      float a = b2[j];
      const float* wr = W2 + j * 2 * HID_;
      for (int k = 0; k < 2 * HID_; ++k) a += wr[k] * hid1[k];
      inp[j] = a;
      inp[HID_ + j] = salf[(size_t)(b * T_ + t) * HID_ + j];
    }
    __syncthreads();
    // LSTM gates (i,f,g,o)
    float g[4];
#pragma unroll
    for (int gi = 0; gi < 4; ++gi) {
      const int row = gi * HID_ + j;
      float acc = bih[row] + bhh[row];
      const float* wi = Wih + row * 2 * HID_;
      for (int k = 0; k < 2 * HID_; ++k) acc += wi[k] * inp[k];
      const float* wh = Whh + row * HID_;
      for (int k = 0; k < HID_; ++k) acc += wh[k] * h[k];
      g[gi] = acc;
    }
    __syncthreads();
    c = sigmoidf_(g[1]) * c + sigmoidf_(g[0]) * tanhf(g[2]);
    h[j] = sigmoidf_(g[3]) * tanhf(c);
    __syncthreads();
    if (j < 3) {
      float d = bo[j];
      const float* wo = Wo + j * HID_;
      for (int k = 0; k < HID_; ++k) d += wo[k] * h[k];
      newp[j] = prev[j] + d;
    }
    __syncthreads();
    if (j == 0) {
      const float n2 = newp[0] * newp[0] + newp[1] * newp[1] + newp[2] * newp[2];
      const float inv = rsqrtf(n2);
      prev[0] = newp[0] * inv;
      prev[1] = newp[1] * inv;
      prev[2] = newp[2] * inv;
    }
    __syncthreads();
    if (j < 3) out[(size_t)(b * T_ + t) * 3 + j] = prev[j];
    __syncthreads();
  }
}

// ---------------------------------------------------------------------------
extern "C" void kernel_launch(void* const* d_in, const int* in_sizes, int n_in,
                              void* d_out, int out_size, void* d_ws, size_t ws_size,
                              hipStream_t stream) {
  const float* pos     = (const float*)d_in[0];
  const float* sal     = (const float*)d_in[1];
  const float* Wih_enc = (const float*)d_in[2];
  const float* Whh_enc = (const float*)d_in[3];
  const float* bih_enc = (const float*)d_in[4];
  const float* bhh_enc = (const float*)d_in[5];
  const float* W1      = (const float*)d_in[6];
  const float* b1      = (const float*)d_in[7];
  const float* W2      = (const float*)d_in[8];
  const float* b2      = (const float*)d_in[9];
  const float* k1      = (const float*)d_in[10];
  const float* bc1     = (const float*)d_in[11];
  const float* k2      = (const float*)d_in[12];
  const float* bc2     = (const float*)d_in[13];
  const float* k3      = (const float*)d_in[14];
  const float* bc3     = (const float*)d_in[15];
  const float* Wf      = (const float*)d_in[16];
  const float* bf      = (const float*)d_in[17];
  const float* Wih_dec = (const float*)d_in[18];
  const float* Whh_dec = (const float*)d_in[19];
  const float* bih_dec = (const float*)d_in[20];
  const float* bhh_dec = (const float*)d_in[21];
  const float* Wo      = (const float*)d_in[22];
  const float* bo      = (const float*)d_in[23];

  char* ws = (char*)d_ws;
  _Float16* WfH  = (_Float16*)(ws + OFF_WFH);
  _Float16* B2   = (_Float16*)(ws + OFF_B2);
  _Float16* B3   = (_Float16*)(ws + OFF_B3);
  _Float16* out3 = (_Float16*)(ws + OFF_OUT3);
  float*    salf = (float*)(ws + OFF_SALF);
  float*    hEnc = (float*)(ws + OFF_HENC);
  float*    cEnc = (float*)(ws + OFF_CENC);

  prep_kernel<<<512, 256, 0, stream>>>(Wf, k2, k3, WfH, B2, B3);
  enc_kernel<<<B_, HID_, 0, stream>>>(pos, Wih_enc, Whh_enc, bih_enc, bhh_enc, hEnc, cEnc);
  cnn_kernel<<<B_ * T_, 256, 0, stream>>>(sal, k1, bc1, bc2, bc3, B2, B3, out3);
  fc_kernel<<<(B_ * T_) / 16, 128, 0, stream>>>(out3, WfH, bf, salf);
  dec_kernel<<<B_, HID_, 0, stream>>>(pos, salf, W1, b1, W2, b2,
                                      Wih_dec, Whh_dec, bih_dec, bhh_dec, Wo, bo,
                                      hEnc, cEnc, (float*)d_out);
}
